// MultiNodeMlp_43181601194349
// MI455X (gfx1250) — matrix operand order repaired and checked by reading, not compiled, not run on hardware
//
#include <hip/hip_runtime.h>
#include <cmath>

// ---------------------------------------------------------------------------
// MultiNodeMlp fused kernel for MI455X (gfx1250, wave32, WMMA, 320KB LDS/WGP)
//
//   sigma_kernel : power iteration -> spectral norm of each of 50 weight mats
//   mlp_fused    : per (node, 128-row block):
//                    GEMM1 (x * W1/s1 + b1, erf GELU) -> h in LDS (f16)
//                    GEMM2 (h * W2/s2 + b2)           -> out (f32)
//   Precision: f16 hi/lo split (3 WMMA products) for GEMM1 operands and for
//   W2 in GEMM2; h kept as single f16 (rel err ~2^-12).
//   Global fp32 tiles are register-pipelined one k-step ahead so load latency
//   overlaps the WMMA block instead of sitting between the two barriers.
// ---------------------------------------------------------------------------

constexpr int Bsz = 4096;
constexpr int Nn  = 25;
constexpr int Dd  = 1024;

typedef __attribute__((ext_vector_type(16))) _Float16 v16h;
typedef __attribute__((ext_vector_type(8)))  _Float16 h8;
typedef __attribute__((ext_vector_type(4)))  _Float16 h4;
typedef __attribute__((ext_vector_type(8)))  float    v8f;

// LDS geometry
constexpr int HSTR   = 1032;                 // h row stride in halves (banks +4/row)
constexpr int TSTR   = 40;                   // staging row stride in halves (80B, 16B-aligned)
constexpr int HBYTES = 128 * HSTR * 2;       // 264192
constexpr int TBYTES = 128 * TSTR * 2;       // 10240 per staging plane
constexpr int SMEM_BYTES = HBYTES + 4 * TBYTES;  // 305152  (<= 320KB)

union Frag { v16h v; h8 h[2]; };

__device__ __forceinline__ v8f wmma_f16(v16h a, v16h b, v8f c) {
  return __builtin_amdgcn_wmma_f32_16x16x32_f16(
      /*neg_a=*/false, a, /*neg_b=*/false, b,
      /*c_mod=*/(short)0, c, /*reuse_a=*/false, /*reuse_b=*/false);
}

// A fragment (16-bit, 16x32): lane<16 -> K0..7 | K16..23 ; lane>=16 -> +8
__device__ __forceinline__ v16h load_afrag(const _Float16* rowp, int sel8) {
  Frag f;
  f.h[0] = *(const h8*)(rowp + sel8);
  f.h[1] = *(const h8*)(rowp + 16 + sel8);
  return f.v;
}
// B fragment (16-bit, 32x16): lane<16 -> K0..7 | K16..23 ; lane>=16 -> +8
__device__ __forceinline__ v16h load_bfrag(const _Float16* colp, int sel16) {
  Frag f;
  f.h[0] = *(const h8*)(colp + sel16 / 2);
  f.h[1] = *(const h8*)(colp + 16 + sel16 / 2);
  return f.v;
}

// ---- register-pipelined staging -------------------------------------------
// Load input tile [128m x 32k] fp32 into 4 float4 regs (coalesced).
__device__ __forceinline__ void load_x_tile(const float* __restrict__ xg,
                                            int tid, float4* r) {
#pragma unroll
  for (int i = 0; i < 4; ++i) {
    int idx = tid + i * 256;
    int row = idx >> 3;
    int kq  = (idx & 7) << 2;
    r[i] = *(const float4*)(xg + (size_t)row * (Nn * Dd) + kq);
  }
  __builtin_prefetch(xg + 64, 0, 1);  // k-step+2 (speculative global_prefetch_b8)
}
// Convert + store regs -> hi/lo f16 planes, row-major stride TSTR.
__device__ __forceinline__ void store_x_tile(const float4* r, int tid,
                                             _Float16* Ahi, _Float16* Alo) {
#pragma unroll
  for (int i = 0; i < 4; ++i) {
    int idx = tid + i * 256;
    int row = idx >> 3;
    int kq  = (idx & 7) << 2;
    const float4 g = r[i];
    _Float16 h0 = (_Float16)g.x, h1 = (_Float16)g.y,
             h2 = (_Float16)g.z, h3 = (_Float16)g.w;
    h4 vh = {h0, h1, h2, h3};
    h4 vl = {(_Float16)(g.x - (float)h0), (_Float16)(g.y - (float)h1),
             (_Float16)(g.z - (float)h2), (_Float16)(g.w - (float)h3)};
    int o = row * TSTR + kq;
    *(h4*)(Ahi + o) = vh;
    *(h4*)(Alo + o) = vl;
  }
}
// Load weight tile [32k x 128n] fp32 into 4 float4 regs (coalesced).
__device__ __forceinline__ void load_w_tile(const float* __restrict__ wg,
                                            int tid, float4* r) {
#pragma unroll
  for (int i = 0; i < 4; ++i) {
    int idx = tid + i * 256;
    int kr  = idx >> 5;
    int nq  = (idx & 31) << 2;
    r[i] = *(const float4*)(wg + (size_t)kr * Dd + nq);
  }
  __builtin_prefetch(wg + (size_t)64 * Dd, 0, 1);  // k-step+2
}
__device__ __forceinline__ void split_store(_Float16* hi, _Float16* lo,
                                            int idx, float f) {
  _Float16 h = (_Float16)f;
  hi[idx] = h;
  lo[idx] = (_Float16)(f - (float)h);
}
// Scale + split + store regs TRANSPOSED [n][k] into hi/lo planes.
__device__ __forceinline__ void store_w_tile(const float4* r, float s, int tid,
                                             _Float16* Bhi, _Float16* Blo) {
#pragma unroll
  for (int i = 0; i < 4; ++i) {
    int idx = tid + i * 256;
    int kr  = idx >> 5;
    int nq  = (idx & 31) << 2;
    const float4 g = r[i];
    split_store(Bhi, Blo, (nq + 0) * TSTR + kr, g.x * s);
    split_store(Bhi, Blo, (nq + 1) * TSTR + kr, g.y * s);
    split_store(Bhi, Blo, (nq + 2) * TSTR + kr, g.z * s);
    split_store(Bhi, Blo, (nq + 3) * TSTR + kr, g.w * s);
  }
}

// Branch-free exact-erf GELU (Abramowitz-Stegun 7.1.26, |err| <= 1.5e-7;
// far below the f16 quantization of h). Keeps EXEC all-1s around WMMA phases.
__device__ __forceinline__ float gelu_erf(float v) {
  const float x  = v * 0.70710678118654752f;
  const float ax = fabsf(x);
  const float t  = __builtin_amdgcn_rcpf(1.0f + 0.3275911f * ax);
  const float p  = t * (0.254829592f +
                   t * (-0.284496736f +
                   t * (1.421413741f +
                   t * (-1.453152027f + t * 1.061405429f))));
  const float e  = p * __expf(-ax * ax);
  const float er = copysignf(1.0f - e, x);
  return 0.5f * v * (1.0f + er);
}

// ---------------------------------------------------------------------------
// Kernel 1: spectral norm via power iteration (one block per matrix, 50 mats)
// ---------------------------------------------------------------------------
__global__ __launch_bounds__(1024) void sigma_kernel(
    const float* __restrict__ w1, const float* __restrict__ w2,
    float* __restrict__ sigma) {
  __shared__ float v[Dd];
  __shared__ float u[Dd];
  __shared__ float red[32];

  const int mat = blockIdx.x;  // 0..24 -> W1, 25..49 -> W2
  const float* W = (mat < 25) ? (w1 + (size_t)mat * Dd * Dd)
                              : (w2 + (size_t)(mat - 25) * Dd * Dd);
  const int tid  = threadIdx.x;
  const int lane = tid & 31;
  const int wid  = tid >> 5;

  v[tid] = 0.03125f;  // 1/sqrt(1024)
  __syncthreads();

  float nrm = 1.0f;
  const int tc = tid & 255;  // column group (4 cols)
  const int tk = tid >> 8;   // k quarter

  for (int it = 0; it < 40; ++it) {
    // u = W v : one wave per row, coalesced float4, shuffle reduce
    for (int r = wid; r < Dd; r += 32) {
      const float4* row = (const float4*)(W + (size_t)r * Dd);
      float s = 0.0f;
      for (int j = lane; j < Dd / 4; j += 32) {
        float4 g  = row[j];
        float4 vv = *(const float4*)(v + j * 4);
        s += g.x * vv.x + g.y * vv.y + g.z * vv.z + g.w * vv.w;
      }
      for (int off = 16; off > 0; off >>= 1) s += __shfl_down(s, off, 32);
      if (lane == 0) u[r] = s;
    }
    __syncthreads();
    v[tid] = 0.0f;
    __syncthreads();
    // v = W^T u : thread owns 4 columns, k split 4 ways, coalesced float4
    float ax = 0.f, ay = 0.f, az = 0.f, aw = 0.f;
    for (int k = tk * 256; k < tk * 256 + 256; ++k) {
      const float uk = u[k];
      const float4 g = *(const float4*)(W + (size_t)k * Dd + tc * 4);
      ax += g.x * uk; ay += g.y * uk; az += g.z * uk; aw += g.w * uk;
    }
    atomicAdd(&v[tc * 4 + 0], ax);
    atomicAdd(&v[tc * 4 + 1], ay);
    atomicAdd(&v[tc * 4 + 2], az);
    atomicAdd(&v[tc * 4 + 3], aw);
    __syncthreads();
    // nrm = ||v||^2 ; normalize
    float s = v[tid] * v[tid];
    for (int off = 16; off > 0; off >>= 1) s += __shfl_down(s, off, 32);
    if (lane == 0) red[wid] = s;
    __syncthreads();
    if (wid == 0) {
      float t = red[lane];
      for (int off = 16; off > 0; off >>= 1) t += __shfl_down(t, off, 32);
      if (lane == 0) red[0] = t;
    }
    __syncthreads();
    nrm = red[0];
    v[tid] *= rsqrtf(nrm);
    __syncthreads();
  }
  // ||W^T W v|| = sigma^2  ->  nrm = sigma^4
  if (tid == 0) sigma[mat] = sqrtf(sqrtf(nrm));
}

// ---------------------------------------------------------------------------
// Kernel 2: fused normalized-MLP. grid = (32 m-blocks, 25 nodes), 256 threads.
// ---------------------------------------------------------------------------
__global__ __launch_bounds__(256) void mlp_fused_kernel(
    const float* __restrict__ x,  const float* __restrict__ w1,
    const float* __restrict__ b1, const float* __restrict__ w2,
    const float* __restrict__ b2, const float* __restrict__ sig,
    float* __restrict__ out) {
  extern __shared__ char smem[];
  _Float16* hbuf = (_Float16*)smem;                  // [128][HSTR]
  _Float16* Ahi  = (_Float16*)(smem + HBYTES);
  _Float16* Alo  = Ahi + 128 * TSTR;
  _Float16* Bhi  = Alo + 128 * TSTR;
  _Float16* Blo  = Bhi + 128 * TSTR;

  const int tid  = threadIdx.x;
  const int lane = tid & 31;
  const int lhi  = lane >> 4;        // 0/1: low/high half of wave
  const int l15  = lane & 15;
  const int wid  = tid >> 5;
  const int wm   = wid & 1;          // 2 wave-rows  (64 m each)
  const int wn   = wid >> 1;         // 4 wave-cols  (32 n each)

  const int    node = blockIdx.y;
  const size_t m0g  = (size_t)blockIdx.x * 128;

  const float inv_s1 = 1.0f / sig[node];
  const float inv_s2 = 1.0f / sig[25 + node];
  const float* w1n = w1 + (size_t)node * Dd * Dd;
  const float* w2n = w2 + (size_t)node * Dd * Dd;
  const float* xn  = x + (m0g * Nn + node) * Dd;

  const v8f vzero = {0.f, 0.f, 0.f, 0.f, 0.f, 0.f, 0.f, 0.f};

  float4 xa[4], wa[4];

  // ----------------- Phase 1: h = GELU(x*W1/s1 + b1) -> LDS ----------------
  for (int nt = 0; nt < 8; ++nt) {
    const int n0 = nt * 128;
    v8f c[4][2];
#pragma unroll
    for (int mf = 0; mf < 4; ++mf)
#pragma unroll
      for (int nf = 0; nf < 2; ++nf) c[mf][nf] = vzero;

    load_x_tile(xn, tid, xa);                    // ks = 0 tiles -> regs
    load_w_tile(w1n + n0, tid, wa);

    for (int ks = 0; ks < 32; ++ks) {
      __syncthreads();                           // prev compute done
      store_x_tile(xa, tid, Ahi, Alo);
      store_w_tile(wa, inv_s1, tid, Bhi, Blo);
      __syncthreads();                           // tiles ready
      if (ks + 1 < 32) {                         // uniform guard
        load_x_tile(xn + (ks + 1) * 32, tid, xa);            // fly during WMMA
        load_w_tile(w1n + (size_t)((ks + 1) * 32) * Dd + n0, tid, wa);
      }

      v16h ah[4], al[4], bh[2], bl[2];
#pragma unroll
      for (int mf = 0; mf < 4; ++mf) {
        const int row = (wm * 64 + mf * 16 + l15) * TSTR;
        ah[mf] = load_afrag(Ahi + row, lhi * 8);
        al[mf] = load_afrag(Alo + row, lhi * 8);
      }
#pragma unroll
      for (int nf = 0; nf < 2; ++nf) {
        const int col = (wn * 32 + nf * 16 + l15) * TSTR;
        bh[nf] = load_bfrag(Bhi + col, lhi * 16);
        bl[nf] = load_bfrag(Blo + col, lhi * 16);
      }
#pragma unroll
      for (int mf = 0; mf < 4; ++mf)
#pragma unroll
        for (int nf = 0; nf < 2; ++nf) {
          c[mf][nf] = wmma_f16(ah[mf], bh[nf], c[mf][nf]);
          c[mf][nf] = wmma_f16(ah[mf], bl[nf], c[mf][nf]);
          c[mf][nf] = wmma_f16(al[mf], bh[nf], c[mf][nf]);
        }
    }
    // epilogue: bias + branch-free erf GELU, write f16 into hbuf
#pragma unroll
    for (int nf = 0; nf < 2; ++nf) {
      const int   nc   = n0 + wn * 32 + nf * 16 + l15;
      const float bias = b1[node * Dd + nc];
#pragma unroll
      for (int mf = 0; mf < 4; ++mf) {
#pragma unroll
        for (int r = 0; r < 8; ++r) {
          const float gg = gelu_erf(c[mf][nf][r] + bias);
          hbuf[(size_t)(wm * 64 + mf * 16 + lhi * 8 + r) * HSTR + nc] =
              (_Float16)gg;
        }
      }
    }
  }
  __syncthreads();

  // ----------------- Phase 2: out = h*W2/s2 + b2 ---------------------------
  for (int ot = 0; ot < 8; ++ot) {
    const int o0 = ot * 128;
    v8f c[4][2];
#pragma unroll
    for (int mf = 0; mf < 4; ++mf)
#pragma unroll
      for (int nf = 0; nf < 2; ++nf) c[mf][nf] = vzero;

    load_w_tile(w2n + o0, tid, wa);              // ks = 0 tile -> regs

    for (int ks = 0; ks < 32; ++ks) {
      __syncthreads();
      store_w_tile(wa, inv_s2, tid, Bhi, Blo);
      __syncthreads();
      if (ks + 1 < 32)                           // uniform guard
        load_w_tile(w2n + (size_t)((ks + 1) * 32) * Dd + o0, tid, wa);

      v16h a[4], bh[2], bl[2];
#pragma unroll
      for (int mf = 0; mf < 4; ++mf)
        a[mf] = load_afrag(
            hbuf + (size_t)(wm * 64 + mf * 16 + l15) * HSTR + ks * 32, lhi * 8);
#pragma unroll
      for (int nf = 0; nf < 2; ++nf) {
        const int col = (wn * 32 + nf * 16 + l15) * TSTR;
        bh[nf] = load_bfrag(Bhi + col, lhi * 16);
        bl[nf] = load_bfrag(Blo + col, lhi * 16);
      }
#pragma unroll
      for (int mf = 0; mf < 4; ++mf)
#pragma unroll
        for (int nf = 0; nf < 2; ++nf) {
          c[mf][nf] = wmma_f16(a[mf], bh[nf], c[mf][nf]);
          c[mf][nf] = wmma_f16(a[mf], bl[nf], c[mf][nf]);
        }
    }
    // epilogue: bias + store f32, coalesced across lanes
#pragma unroll
    for (int nf = 0; nf < 2; ++nf) {
      const int   oc   = o0 + wn * 32 + nf * 16 + l15;
      const float bias = b2[node * Dd + oc];
#pragma unroll
      for (int mf = 0; mf < 4; ++mf) {
#pragma unroll
        for (int r = 0; r < 8; ++r) {
          const size_t m = m0g + (size_t)(wm * 64 + mf * 16 + lhi * 8 + r);
          out[(m * Nn + node) * Dd + oc] = c[mf][nf][r] + bias;
        }
      }
    }
  }
}

// ---------------------------------------------------------------------------
extern "C" void kernel_launch(void* const* d_in, const int* in_sizes, int n_in,
                              void* d_out, int out_size, void* d_ws,
                              size_t ws_size, hipStream_t stream) {
  (void)in_sizes; (void)n_in; (void)out_size; (void)ws_size;
  const float* x  = (const float*)d_in[0];
  const float* w1 = (const float*)d_in[1];
  const float* b1 = (const float*)d_in[2];
  const float* w2 = (const float*)d_in[3];
  const float* b2 = (const float*)d_in[4];
  float* sig = (float*)d_ws;   // 50 floats
  float* out = (float*)d_out;

  hipFuncSetAttribute((const void*)mlp_fused_kernel,
                      hipFuncAttributeMaxDynamicSharedMemorySize, SMEM_BYTES);

  sigma_kernel<<<50, 1024, 0, stream>>>(w1, w2, sig);
  mlp_fused_kernel<<<dim3(Bsz / 128, Nn), 256, SMEM_BYTES, stream>>>(
      x, w1, b1, w2, b2, sig, out);
}
